// MultiHeadSelfAttention_1365799600597
// MI455X (gfx1250) — compile-verified, matrix operand order checked by reading, not run on hardware
//
#include <hip/hip_runtime.h>
#include <hip/hip_bf16.h>
#include <math.h>

// ---------------------------------------------------------------------------
// MI455X (gfx1250) multi-head self-attention, bf16 WMMA pipeline.
//   stage 0: f32 -> bf16 convert (x, Wq, Wk, Wv, Wo)
//   stage 1: QKV projection GEMMs  (async-LDS staged, v_wmma_f32_16x16x32_bf16)
//   stage 2: causal flash attention (wmma scores + online softmax + wmma PV)
//   stage 3: output projection GEMM -> fp32 d_out (same async-staged GEMM)
// ---------------------------------------------------------------------------

typedef __bf16 bf16_t;
typedef __attribute__((ext_vector_type(16))) __bf16 v16bf;
typedef __attribute__((ext_vector_type(8)))  float  v8f;

#define D_MODEL 1024
#define NUM_HEADS 16
#define D_K 64
#define BATCH 2
#define SEQ 2048
#define M_TOTAL (BATCH * SEQ) /* 4096 */
#define LDS_LD 40             /* 32 cols + 8 pad elems (80B row) vs bank stride */

__device__ __forceinline__ bf16_t f2bf(float f) {
  union { float f; unsigned u; } v; v.f = f;
  unsigned u = v.u;
  unsigned r = u + 0x7FFFu + ((u >> 16) & 1u);   // round-to-nearest-even
  union { unsigned short s; bf16_t b; } o;
  o.s = (unsigned short)(r >> 16);
  return o.b;
}

// Load a 16x32 bf16 fragment per ISA 7.12.2 (16-bit A-matrix layout).
// "row" dim -> lane&15 ; half = lane>>4 selects k-group.
//   VGPR v in 0..3 : k = half*8 + 2v + {0,1}
//   VGPR v in 4..7 : k = 16 + half*8 + 2(v-4) + {0,1}
// B-matrix layout mirrors this with N = lane&15, so the same loader serves
// B fragments of row-major [N,K] matrices.
__device__ __forceinline__ v16bf load_frag(const bf16_t* __restrict__ src,
                                           int ld, int r0, int k0) {
  int lane = threadIdx.x & 31;
  int half = lane >> 4;
  int m = lane & 15;
  const bf16_t* p = src + (size_t)(r0 + m) * ld + k0;
  v16bf a;
#pragma unroll
  for (int v = 0; v < 4; ++v) {
    int kk = half * 8 + 2 * v;
    a[2 * v]     = p[kk];
    a[2 * v + 1] = p[kk + 1];
  }
#pragma unroll
  for (int v = 4; v < 8; ++v) {
    int kk = 16 + half * 8 + 2 * (v - 4);
    a[2 * v]     = p[kk];
    a[2 * v + 1] = p[kk + 1];
  }
  return a;
}

// Async DMA: 16B global -> LDS per lane, tracked by ASYNCcnt (ISA §10 / §15.18).
// LDS dest = low 32 bits of the flat shared address (LDS aperture offset).
__device__ __forceinline__ void async_load_b128(const bf16_t* g, bf16_t* l) {
  unsigned loff = (unsigned)(uintptr_t)l;
  asm volatile("global_load_async_to_lds_b128 %0, %1, off"
               :: "v"(loff), "v"(g)
               : "memory");
}

// ---------------------------------------------------------------------------
// Stage 0: elementwise f32 -> bf16
// ---------------------------------------------------------------------------
__global__ void cvt_f32_bf16_kernel(const float* __restrict__ in,
                                    bf16_t* __restrict__ out, int n) {
  int i = blockIdx.x * blockDim.x + threadIdx.x;
  int stride = gridDim.x * blockDim.x;
  for (; i < n; i += stride) out[i] = f2bf(in[i]);
}

// ---------------------------------------------------------------------------
// Stages 1 & 3: C = A @ B^T with async-LDS double-buffered 128x128 block tile.
//   A [4096,1024] bf16 row-major, B [1024,1024] bf16 row-major (rows = N dim).
//   Block = 256 threads = 8 waves; wave w computes rows [w*16,w*16+16) x 128
//   cols (8 fp32 16x16 accumulators). Tiles (128x32) staged via
//   global_load_async_to_lds_b128, overlap via s_wait_asynccnt 4.
// Store modes: 0 -> bf16 [b,h,s,dk] (Q/K); 2 -> bf16 [b,h,dk,s] (V^T);
//              3 -> fp32 row-major [4096,1024] (final output).
// ---------------------------------------------------------------------------
__global__ void gemm_bf16_kernel(const bf16_t* __restrict__ A,
                                 const bf16_t* __restrict__ Bm,
                                 void* __restrict__ outp, int mode) {
  __shared__ bf16_t As[2][128 * LDS_LD];
  __shared__ bf16_t Bs[2][128 * LDS_LD];

  int t = threadIdx.x;
  int wave = t >> 5;
  int lane = t & 31;
  int m0 = blockIdx.y * 128;
  int n0 = blockIdx.x * 128;

  // Stage a 128x32 A tile and 128x32 B tile into LDS buffer `buf` at k-offset
  // `k`: 512 chunks of 16B each (4 chunks per 64B row), 2 chunks per thread
  // per matrix -> 4 async instructions per wave per stage.
  auto stage = [&](int buf, int k) {
#pragma unroll
    for (int j = 0; j < 2; ++j) {
      int c = t + 256 * j;
      int row = c >> 2;
      int col8 = (c & 3) * 8;                       // bf16 elems, 16B chunks
      async_load_b128(A  + (size_t)(m0 + row) * D_MODEL + k + col8,
                      &As[buf][row * LDS_LD + col8]);
      async_load_b128(Bm + (size_t)(n0 + row) * D_MODEL + k + col8,
                      &Bs[buf][row * LDS_LD + col8]);
    }
  };

  v8f acc[8] = {};

  stage(0, 0);
  for (int ks = 0; ks < D_MODEL / 32; ++ks) {
    int cur = ks & 1;
    if (ks + 1 < D_MODEL / 32) {
      stage(cur ^ 1, (ks + 1) * 32);
      // wait for current tile only; keep the 4 just-issued loads in flight
      asm volatile("s_wait_asynccnt 0x4" ::: "memory");
    } else {
      asm volatile("s_wait_asynccnt 0x0" ::: "memory");
    }
    __syncthreads();                 // current tile visible to all waves

    v16bf a = load_frag(&As[cur][0], LDS_LD, wave * 16, 0);
#pragma unroll
    for (int nt = 0; nt < 8; ++nt) {
      v16bf b = load_frag(&Bs[cur][0], LDS_LD, nt * 16, 0);
      acc[nt] = __builtin_amdgcn_wmma_f32_16x16x32_bf16(
          false, a, false, b, (short)0, acc[nt], false, false);
    }
    __syncthreads();                 // all waves done reading before overwrite
  }

  // ---- store (C layout: M = i + 8*half, N = lane&15)
  int half = lane >> 4;
  int n = lane & 15;
  int mrow0 = m0 + wave * 16;
#pragma unroll
  for (int nt = 0; nt < 8; ++nt) {
    int col = n0 + nt * 16 + n;
#pragma unroll
    for (int i = 0; i < 8; ++i) {
      int mg = mrow0 + i + 8 * half;
      float v = acc[nt][i];
      if (mode == 3) {
        ((float*)outp)[(size_t)mg * D_MODEL + col] = v;
      } else {
        bf16_t bv = f2bf(v);
        int b = mg >> 11;             // /SEQ; 16-row tiles never straddle batch
        int srow = mg & (SEQ - 1);
        int h = col >> 6;
        int dk = col & 63;
        if (mode == 2) {
          ((bf16_t*)outp)[(((size_t)b * NUM_HEADS + h) * D_K + dk) * SEQ + srow] = bv;
        } else {
          ((bf16_t*)outp)[(((size_t)b * NUM_HEADS + h) * SEQ + srow) * D_K + dk] = bv;
        }
      }
    }
  }
}

// ---------------------------------------------------------------------------
// Stage 2: causal flash attention.
//   Q,K [b,h,s,dk] bf16 ; Vt [b,h,dk,s] bf16 ; out [b, s, h*dk] bf16.
// Block = 256 threads = 8 waves; wave owns 16 q rows; streams 32-key blocks,
// truncated by the causal bound. Online softmax in fp32 C-layout; P is
// re-laid out to an A-fragment through per-wave LDS (ds stores/loads +
// s_wait_dscnt, wave-coherent so no block barrier is needed).
// ---------------------------------------------------------------------------
__global__ void flash_attn_kernel(const bf16_t* __restrict__ Qb,
                                  const bf16_t* __restrict__ Kb,
                                  const bf16_t* __restrict__ Vtb,
                                  bf16_t* __restrict__ Ob) {
  __shared__ bf16_t lds_p[8][16 * 32];
  int wave = threadIdx.x >> 5;
  int lane = threadIdx.x & 31;
  int half = lane >> 4;
  int n = lane & 15;

  int bh = blockIdx.y;          // 0..31
  int b = bh >> 4, h = bh & 15;
  int q0 = blockIdx.x * 128 + wave * 16;

  const bf16_t* Qh = Qb  + (size_t)bh * SEQ * D_K;
  const bf16_t* Kh = Kb  + (size_t)bh * SEQ * D_K;
  const bf16_t* Vh = Vtb + (size_t)bh * D_K * SEQ;

  v16bf qa0 = load_frag(Qh, D_K, q0, 0);
  v16bf qa1 = load_frag(Qh, D_K, q0, 32);

  v8f o[4] = {};
  float mrow[8], lrow[8];
#pragma unroll
  for (int i = 0; i < 8; ++i) { mrow[i] = -1e30f; lrow[i] = 0.0f; }

  const float scale = 0.125f;   // 1/sqrt(64)

  for (int s0 = 0; s0 < q0 + 16; s0 += 32) {
    // ---- scores for 32 keys: two 16x16 tiles, dk contraction = 2 wmma each
    v8f sacc[2];
#pragma unroll
    for (int t = 0; t < 2; ++t) {
      v8f c = {};
      v16bf kb0 = load_frag(Kh, D_K, s0 + t * 16, 0);
      c = __builtin_amdgcn_wmma_f32_16x16x32_bf16(false, qa0, false, kb0,
                                                  (short)0, c, false, false);
      v16bf kb1 = load_frag(Kh, D_K, s0 + t * 16, 32);
      c = __builtin_amdgcn_wmma_f32_16x16x32_bf16(false, qa1, false, kb1,
                                                  (short)0, c, false, false);
      sacc[t] = c;
    }

    // ---- scale + causal mask + running max (row reduce over 16-lane half)
    float mnew[8];
#pragma unroll
    for (int i = 0; i < 8; ++i) {
      int qi = q0 + i + 8 * half;
      float s0v = sacc[0][i] * scale;
      float s1v = sacc[1][i] * scale;
      if (s0 + n > qi)      s0v = -1e30f;
      if (s0 + 16 + n > qi) s1v = -1e30f;
      sacc[0][i] = s0v;
      sacc[1][i] = s1v;
      float rm = fmaxf(s0v, s1v);
#pragma unroll
      for (int mk = 1; mk < 16; mk <<= 1)
        rm = fmaxf(rm, __shfl_xor(rm, mk, 32));
      mnew[i] = fmaxf(mrow[i], rm);
    }

    // ---- exp, row sum, online rescale of O and l
#pragma unroll
    for (int i = 0; i < 8; ++i) {
      float p0 = __expf(sacc[0][i] - mnew[i]);
      float p1 = __expf(sacc[1][i] - mnew[i]);
      sacc[0][i] = p0;
      sacc[1][i] = p1;
      float rs = p0 + p1;
#pragma unroll
      for (int mk = 1; mk < 16; mk <<= 1)
        rs += __shfl_xor(rs, mk, 32);
      float alpha = __expf(mrow[i] - mnew[i]);
      lrow[i] = lrow[i] * alpha + rs;
      mrow[i] = mnew[i];
#pragma unroll
      for (int nt = 0; nt < 4; ++nt) o[nt][i] *= alpha;
    }

    // ---- C-layout P -> row-major 16x32 in LDS -> A-fragment
#pragma unroll
    for (int t = 0; t < 2; ++t)
#pragma unroll
      for (int i = 0; i < 8; ++i) {
        int row = i + 8 * half;
        int col = t * 16 + n;
        lds_p[wave][row * 32 + col] = f2bf(sacc[t][i]);
      }
    asm volatile("s_wait_dscnt 0" ::: "memory");
    v16bf pfrag = load_frag(&lds_p[wave][0], 32, 0, 0);

    // ---- O(16x64) += P(16x32) @ V(32x64)   (Vt gives contiguous-k B frags)
#pragma unroll
    for (int nt = 0; nt < 4; ++nt) {
      v16bf vb = load_frag(Vh, SEQ, nt * 16, s0);
      o[nt] = __builtin_amdgcn_wmma_f32_16x16x32_bf16(
          false, pfrag, false, vb, (short)0, o[nt], false, false);
    }
  }

  // ---- normalize and store to [b, s, h*dk] (head-concatenated, bf16)
#pragma unroll
  for (int nt = 0; nt < 4; ++nt)
#pragma unroll
    for (int i = 0; i < 8; ++i) {
      int srow = q0 + i + 8 * half;
      int col = h * D_K + nt * 16 + n;
      float val = o[nt][i] / lrow[i];
      Ob[((size_t)b * SEQ + srow) * D_MODEL + col] = f2bf(val);
    }
}

// ---------------------------------------------------------------------------
// Host launcher
// ---------------------------------------------------------------------------
extern "C" void kernel_launch(void* const* d_in, const int* in_sizes, int n_in,
                              void* d_out, int out_size, void* d_ws,
                              size_t ws_size, hipStream_t stream) {
  const float* x  = (const float*)d_in[0];  // [2,2048,1024]
  const float* wq = (const float*)d_in[1];  // [16,64,1024]
  const float* wk = (const float*)d_in[2];
  const float* wv = (const float*)d_in[3];
  const float* wo = (const float*)d_in[4];  // [1024,1024]
  float* out = (float*)d_out;               // [2,2048,1024]

  char* ws = (char*)d_ws;
  size_t off = 0;
  auto wsalloc = [&](size_t bytes) -> void* {
    void* p = ws + off;
    off += (bytes + 255) & ~(size_t)255;
    return p;
  };

  const size_t nx = (size_t)M_TOTAL * D_MODEL;     // 4,194,304
  const size_t nw = (size_t)D_MODEL * D_MODEL;     // 1,048,576

  bf16_t* xb   = (bf16_t*)wsalloc(nx * 2);
  bf16_t* wqb  = (bf16_t*)wsalloc(nw * 2);
  bf16_t* wkb  = (bf16_t*)wsalloc(nw * 2);
  bf16_t* wvb  = (bf16_t*)wsalloc(nw * 2);
  bf16_t* wob  = (bf16_t*)wsalloc(nw * 2);
  bf16_t* Qb   = (bf16_t*)wsalloc(nx * 2);         // [b,h,s,dk]
  bf16_t* Kbuf = (bf16_t*)wsalloc(nx * 2);         // [b,h,s,dk]
  bf16_t* Vtb  = (bf16_t*)wsalloc(nx * 2);         // [b,h,dk,s]
  bf16_t* Attn = (bf16_t*)wsalloc(nx * 2);         // [b,s,h*dk]

  // stage 0: conversions
  cvt_f32_bf16_kernel<<<1024, 256, 0, stream>>>(x,  xb,  (int)nx);
  cvt_f32_bf16_kernel<<<512,  256, 0, stream>>>(wq, wqb, (int)nw);
  cvt_f32_bf16_kernel<<<512,  256, 0, stream>>>(wk, wkb, (int)nw);
  cvt_f32_bf16_kernel<<<512,  256, 0, stream>>>(wv, wvb, (int)nw);
  cvt_f32_bf16_kernel<<<512,  256, 0, stream>>>(wo, wob, (int)nw);

  // stages 1 & 3 grid: 8 col-blocks x 32 row-blocks of 128x128
  dim3 gg(D_MODEL / 128, M_TOTAL / 128);
  gemm_bf16_kernel<<<gg, 256, 0, stream>>>(xb, wqb, Qb,   0);
  gemm_bf16_kernel<<<gg, 256, 0, stream>>>(xb, wkb, Kbuf, 0);
  gemm_bf16_kernel<<<gg, 256, 0, stream>>>(xb, wvb, Vtb,  2);

  // stage 2: flash attention (grid: 16 q-tiles x 32 (b,h) pairs)
  dim3 ag(SEQ / 128, BATCH * NUM_HEADS);
  flash_attn_kernel<<<ag, 256, 0, stream>>>(Qb, Kbuf, Vtb, Attn);

  // stage 3: output projection -> fp32 d_out
  gemm_bf16_kernel<<<gg, 256, 0, stream>>>(Attn, wob, out, 3);
}
